// Conv1_14654428414366
// MI455X (gfx1250) — matrix operand order repaired and checked by reading, not compiled, not run on hardware
//
#include <hip/hip_runtime.h>

typedef __attribute__((ext_vector_type(16))) _Float16 v16h;
typedef __attribute__((ext_vector_type(8)))  _Float16 v8h;
typedef __attribute__((ext_vector_type(8)))  float    v8f;
typedef __attribute__((ext_vector_type(4)))  unsigned int v4u;
typedef __attribute__((ext_vector_type(8)))  int v8i;
typedef __attribute__((ext_vector_type(4)))  int v4i;

#define D_IN   256
#define D_OUT  64
#define NSPLIT 4
#define CHUNK  128           // N columns per LDS-staged chunk
#define ROWS_PER_BLOCK 128   // 8 waves x 16 rows

__constant__ float kInvT = 1.0f / 0.07f;

__device__ __forceinline__ v8f wmma_f16(v16h a, v16h b, v8f c) {
  return __builtin_amdgcn_wmma_f32_16x16x32_f16(false, a, false, b, (short)0, c,
                                                false, false);
}

// A-operand gather: lane holds row m, halves K={off..off+7} and {off+8..off+15}
__device__ __forceinline__ v16h ld_a_pair(const _Float16* p) {
  v8h lo = *(const v8h*)(p);
  v8h hi = *(const v8h*)(p + 8);
  v16h r;
#pragma unroll
  for (int i = 0; i < 8; ++i) { r[i] = lo[i]; r[i + 8] = hi[i]; }
  return r;
}

// B-operand: 16 contiguous halves (32B aligned) -> 2x ds_load_b128 / global b128
__device__ __forceinline__ v16h ld16(const _Float16* p) {
  return *(const v16h*)p;
}

// Generic LDS pointer -> LDS byte address (low 32 bits hold the LDS offset)
__device__ __forceinline__ unsigned lds_addr(const void* p) {
  return (unsigned)(uintptr_t)p;
}

// ---- Tensor Data Mover: 2D tile load, Global -> LDS (D# per ISA ch.8) ----
// tensor dims measured from the tile start; TDM zero-fills OOB reads, which
// implements our row/col tail handling for free.
__device__ __forceinline__ void tdm_load_2d(unsigned ldsOff, const void* gptr,
                                            unsigned dataSizeCode,  // 0=1B,1=2B
                                            unsigned tile0, unsigned tile1,
                                            unsigned td0, unsigned td1,
                                            unsigned long long stride0) {
  unsigned long long ga = (unsigned long long)(uintptr_t)gptr;
  v4u g0;
  g0[0] = 1u;                                   // count=1, user mode
  g0[1] = ldsOff;                               // lds_addr
  g0[2] = (unsigned)ga;                         // global_addr[31:0]
  g0[3] = (unsigned)((ga >> 32) & 0x1FFFFFFu) | (2u << 30);  // addr[56:32]|type=2
  v8i g1;
  g1[0] = (int)(dataSizeCode << 16);            // wg_mask=0, data_size, no flags
  g1[1] = (int)((td0 & 0xFFFFu) << 16);         // tensor_dim0[15:0]
  g1[2] = (int)((td0 >> 16) | ((td1 & 0xFFFFu) << 16));      // td0 hi | td1 lo
  g1[3] = (int)((td1 >> 16) | (tile0 << 16));   // td1 hi | tile_dim0
  g1[4] = (int)(tile1 & 0xFFFFu);               // tile_dim1 (tile_dim2 = 0)
  g1[5] = (int)(stride0 & 0xFFFFFFFFull);       // tensor_dim0_stride[31:0]
  g1[6] = (int)((stride0 >> 32) & 0xFFFFull);   // stride0[47:32] (stride1 = 0)
  g1[7] = 0;
  v4i z = {0, 0, 0, 0};
#if __clang_major__ >= 23
  v8i z8 = {0, 0, 0, 0, 0, 0, 0, 0};
  __builtin_amdgcn_tensor_load_to_lds(g0, g1, z, z, z8, 0);
#else
  __builtin_amdgcn_tensor_load_to_lds(g0, g1, z, z, 0);
#endif
}

// ---------------------------------------------------------------- kernel 1
__global__ void cvt_x_kernel(const float* __restrict__ x,
                             _Float16* __restrict__ xh, int n) {
  int i = blockIdx.x * blockDim.x + threadIdx.x;
  if (i < n) xh[i] = (_Float16)x[i];
}

// ---------------------------------------------------------------- kernel 2
// projH = f16(anchor @ wt), anchor [Na][256] f32, wt [256][64] f32
__global__ void proj_kernel(const float* __restrict__ anchor,
                            const float* __restrict__ wt,
                            _Float16* __restrict__ projH, int Na) {
  __shared__ _Float16 sWtT[D_OUT * D_IN];  // transposed wt, f16, 32 KB

  const int tid = threadIdx.x;
  {
    const float* wrow = wt + (size_t)tid * D_OUT;
#pragma unroll 4
    for (int j = 0; j < D_OUT; ++j)
      sWtT[j * D_IN + tid] = (_Float16)wrow[j];
  }
  __syncthreads();

  const int lane = tid & 31, wave = tid >> 5;
  const int lm = lane & 15, lhi = lane >> 4;
  const long rowBase = (long)blockIdx.x * ROWS_PER_BLOCK + wave * 16;

  long rg = rowBase + lm;
  if (rg >= Na) rg = Na - 1;
  const float* arow = anchor + rg * D_IN;

  v8f o[4] = {};
#pragma unroll
  for (int s = 0; s < 8; ++s) {          // K slices of 32
    const int k0 = s * 32, off = lhi * 16;
    v8f f0 = *(const v8f*)(arow + k0 + off);
    v8f f1 = *(const v8f*)(arow + k0 + 8 + off);
    v16h a;
#pragma unroll
    for (int i = 0; i < 8; ++i) { a[i] = (_Float16)f0[i]; a[i + 8] = (_Float16)f1[i]; }
#pragma unroll
    for (int c = 0; c < 4; ++c) {
      v16h b = ld16(&sWtT[(c * 16 + lm) * D_IN + k0 + lhi * 16]);
      o[c] = wmma_f16(a, b, o[c]);
    }
  }
#pragma unroll
  for (int r = 0; r < 8; ++r) {
    const long row = rowBase + r + lhi * 8;
    if (row < Na) {
#pragma unroll
      for (int c = 0; c < 4; ++c)
        projH[row * D_OUT + c * 16 + lm] = (_Float16)o[c][r];
    }
  }
}

// ---------------------------------------------------------------- kernel 3
// Fused masked-softmax attention, no running max (|s|/T << 88).
// TDM double-buffers the x chunk + adjacency tile under the WMMA compute.
__global__ void attn_kernel(const _Float16* __restrict__ xh,    // [N][64]
                            const _Float16* __restrict__ projH, // [Na][64]
                            const unsigned char* __restrict__ adjs,
                            const int* __restrict__ idxp,
                            float* __restrict__ partO,          // [NSPLIT][Na][64]
                            float* __restrict__ partS,          // [NSPLIT][Na]
                            int N, int Na) {
  __shared__ _Float16 sXh[2][CHUNK * D_OUT];               // 2 x 16 KB
  __shared__ _Float16 sXhT[D_OUT * CHUNK];                 // 16 KB (per-iter)
  __shared__ unsigned char sAdj[2][ROWS_PER_BLOCK * CHUNK];// 2 x 16 KB
  __shared__ _Float16 sProbs[8][16 * CHUNK];               // 32 KB

  const int tid = threadIdx.x;
  const int lane = tid & 31, wave = tid >> 5;
  const int lm = lane & 15, lhi = lane >> 4;
  const int split = blockIdx.y;
  const long rowBlock = (long)blockIdx.x * ROWS_PER_BLOCK;
  const long rowBase = rowBlock + wave * 16;

  const int id = __builtin_amdgcn_readfirstlane(*idxp);
  const unsigned char* adj = adjs + (size_t)id * (size_t)Na * (size_t)N;

  // P tile A-operands (K=64 -> two slices), resident for whole kernel
  long prg = rowBase + lm;
  if (prg >= Na) prg = Na - 1;
  const _Float16* prow = projH + prg * D_OUT;
  const int aoff = lhi * 16;
  const v16h pa0 = ld_a_pair(prow + aoff);        // K 0..31
  const v16h pa1 = ld_a_pair(prow + 32 + aoff);   // K 32..63

  v8f o[4] = {};
  float rs[8] = {0, 0, 0, 0, 0, 0, 0, 0};

  const int nChunks = (N + CHUNK - 1) / CHUNK;

  // prologue: wave 0 issues TDM loads for this split's first chunk
  if (wave == 0 && split < nChunks) {
    const int n0 = split * CHUNK;
    tdm_load_2d(lds_addr(&sXh[0][0]), xh + (size_t)n0 * D_OUT,
                /*2B*/ 1, D_OUT, CHUNK, D_OUT, (unsigned)(N - n0), D_OUT);
    tdm_load_2d(lds_addr(&sAdj[0][0]), adj + (size_t)rowBlock * N + n0,
                /*1B*/ 0, CHUNK, ROWS_PER_BLOCK,
                (unsigned)(N - n0), (unsigned)(Na - rowBlock),
                (unsigned long long)N);
  }

  int p = 0;
  for (int ch = split; ch < nChunks; ch += NSPLIT, p ^= 1) {
    __builtin_amdgcn_s_wait_tensorcnt(0);   // wave0's pair for chunk ch done
    __syncthreads();                        // buf[p] valid for all waves

    // issue next chunk into the other buffer (overlaps with compute below)
    if (wave == 0 && ch + NSPLIT < nChunks) {
      const int n1 = (ch + NSPLIT) * CHUNK;
      tdm_load_2d(lds_addr(&sXh[p ^ 1][0]), xh + (size_t)n1 * D_OUT,
                  1, D_OUT, CHUNK, D_OUT, (unsigned)(N - n1), D_OUT);
      tdm_load_2d(lds_addr(&sAdj[p ^ 1][0]), adj + (size_t)rowBlock * N + n1,
                  0, CHUNK, ROWS_PER_BLOCK,
                  (unsigned)(N - n1), (unsigned)(Na - rowBlock),
                  (unsigned long long)N);
    }

    // build transposed x tile from sXh[p]: thread t -> row t/2, 32-half chunk
    {
      const int n = tid >> 1, hoff = (tid & 1) * 32;
      const v8h* src = (const v8h*)&sXh[p][n * D_OUT + hoff];
#pragma unroll
      for (int i = 0; i < 4; ++i) {
        v8h v = src[i];
#pragma unroll
        for (int j = 0; j < 8; ++j)
          sXhT[(hoff + i * 8 + j) * CHUNK + n] = v[j];
      }
    }
    __syncthreads();

    _Float16* myProbs = sProbs[wave];

    // ---- scores: S(16 x 128) = P(16x64) @ Xt(64x128), via 8 WMMA col-tiles
    for (int t = 0; t < 8; ++t) {
      const _Float16* brow = &sXh[p][(t * 16 + lm) * D_OUT];
      v16h b0 = ld16(brow + lhi * 16);        // K 0..31 slice
      v16h b1 = ld16(brow + 32 + lhi * 16);   // K 32..63 slice
      v8f c = {};
      c = wmma_f16(pa0, b0, c);
      c = wmma_f16(pa1, b1, c);

      const int col = t * 16 + lm;
#pragma unroll
      for (int r = 0; r < 8; ++r) {
        const int m = r + lhi * 8;
        unsigned char ab = sAdj[p][(wave * 16 + m) * CHUNK + col];
        float pr = ab ? __expf(c[r] * kInvT) : 0.0f;
        rs[r] += pr;
        myProbs[m * CHUNK + col] = (_Float16)pr;
      }
    }

    // ---- output: O(16x64) += Pexp(16x128) @ X(128x64), 4 K-chunks of 32
    for (int kk = 0; kk < 4; ++kk) {
      const _Float16* par = &myProbs[lm * CHUNK + kk * 32 + aoff];
      v16h pa = ld_a_pair(par);
#pragma unroll
      for (int c = 0; c < 4; ++c) {
        v16h b = ld16(&sXhT[(c * 16 + lm) * CHUNK + kk * 32 + lhi * 16]);
        o[c] = wmma_f16(pa, b, o[c]);
      }
    }
  }

  // ---- reduce row sums across the 16 lanes of each half-wave
#pragma unroll
  for (int r = 0; r < 8; ++r) {
    float s = rs[r];
    s += __shfl_xor(s, 1, 32);
    s += __shfl_xor(s, 2, 32);
    s += __shfl_xor(s, 4, 32);
    s += __shfl_xor(s, 8, 32);
    rs[r] = s;
  }

  float* po = partO + (size_t)split * (size_t)Na * D_OUT;
  float* ps = partS + (size_t)split * (size_t)Na;
#pragma unroll
  for (int r = 0; r < 8; ++r) {
    const long row = rowBase + r + lhi * 8;
    if (row < Na) {
#pragma unroll
      for (int c = 0; c < 4; ++c) po[row * D_OUT + c * 16 + lm] = o[c][r];
      if (lm == 0) ps[row] = rs[r];
    }
  }
}

// ---------------------------------------------------------------- kernel 4
__global__ void finalize_kernel(const float* __restrict__ partO,
                                const float* __restrict__ partS,
                                const float* __restrict__ weight,
                                const int* __restrict__ idxp,
                                float* __restrict__ out, int Na) {
  int i = blockIdx.x * blockDim.x + threadIdx.x;
  if (i < Na * D_OUT) {
    int row = i >> 6;
    float os = 0.f, ss = 0.f;
#pragma unroll
    for (int s = 0; s < NSPLIT; ++s) {
      os += partO[(size_t)s * Na * D_OUT + i];
      ss += partS[(size_t)s * Na + row];
    }
    out[i] = weight[*idxp] * os / ss;
  }
}

extern "C" void kernel_launch(void* const* d_in, const int* in_sizes, int n_in,
                              void* d_out, int out_size, void* d_ws, size_t ws_size,
                              hipStream_t stream) {
  const float* x = (const float*)d_in[0];
  const float* weight = (const float*)d_in[1];
  const unsigned char* adjs = (const unsigned char*)d_in[2];
  const int* idx = (const int*)d_in[3];
  const float* anchor = (const float*)d_in[4];
  const float* wt = (const float*)d_in[5];
  float* out = (float*)d_out;

  const int N = in_sizes[0] / D_OUT;    // x:      [N, 64]
  const int Na = in_sizes[4] / D_IN;    // anchor: [Na, 256]

  // workspace layout
  _Float16* xhG = (_Float16*)d_ws;                       // N*64 f16
  _Float16* projH = xhG + (size_t)N * D_OUT;             // Na*64 f16
  float* partO = (float*)(projH + (size_t)Na * D_OUT);   // NSPLIT*Na*64 f32
  float* partS = partO + (size_t)NSPLIT * Na * D_OUT;    // NSPLIT*Na f32

  cvt_x_kernel<<<(N * D_OUT + 255) / 256, 256, 0, stream>>>(x, xhG, N * D_OUT);

  const int rowBlocks = (Na + ROWS_PER_BLOCK - 1) / ROWS_PER_BLOCK;
  proj_kernel<<<rowBlocks, 256, 0, stream>>>(anchor, wt, projH, Na);

  attn_kernel<<<dim3(rowBlocks, NSPLIT), 256, 0, stream>>>(
      xhG, projH, adjs, idx, partO, partS, N, Na);

  finalize_kernel<<<(Na * D_OUT + 255) / 256, 256, 0, stream>>>(
      partO, partS, weight, idx, out, Na);
}
